// GAT_81784767250897
// MI455X (gfx1250) — matrix operand order repaired and checked by reading, not compiled, not run on hardware
//
#include <hip/hip_runtime.h>

// GAT 2-layer forward for MI455X (gfx1250, wave32, WMMA).
// Attention aggregation is flash-attention style with v_wmma_f32_16x16x32_bf16:
// A = exp-weights built in-register per lane in the native 16x32 bf16 A layout,
// B = feature tile staged in LDS as bf16, C = f32 accumulators.
// adj tiles staged with GLOBAL_LOAD_ASYNC_TO_LDS_B128 when available.

#define N_NODES 4096
#define F_IN    128
#define U0      16
#define H0      4

typedef __attribute__((ext_vector_type(16))) __bf16 v16bf;
typedef __attribute__((ext_vector_type(2)))  __bf16 v2bf;
typedef __attribute__((ext_vector_type(8)))  float  v8f;

union FragBF { v16bf v; unsigned u[8]; };

// native f32->bf16 (RNE) pack; compiler lowers fptrunc to gfx1250 bf16 cvts
__device__ __forceinline__ unsigned pack2bf(float lo, float hi) {
    union { v2bf v; unsigned u; } c;
    c.v[0] = (__bf16)lo;
    c.v[1] = (__bf16)hi;
    return c.u;
}

#if __has_builtin(__builtin_amdgcn_global_load_async_to_lds_b128)
#define HAVE_ASYNC_LDS 1
#else
#define HAVE_ASYNC_LDS 0
#endif

#if HAVE_ASYNC_LDS
// exact pointee type from the builtin's signature: vector_size(16) int
typedef int v4i_t __attribute__((vector_size(16)));
typedef __attribute__((address_space(1))) v4i_t* async_gp_t;
typedef __attribute__((address_space(3))) v4i_t* async_lp_t;
#endif

// stage 16 bytes global -> LDS (async direct-to-LDS when available)
__device__ __forceinline__ void stage16(const float* g, float* l) {
#if HAVE_ASYNC_LDS
    __builtin_amdgcn_global_load_async_to_lds_b128(
        (async_gp_t)(unsigned long long)(const void*)g,
        (async_lp_t)(unsigned)(unsigned long long)(const void*)l,
        0, 0);
#else
    *(float4*)l = *(const float4*)g;
#endif
}

__device__ __forceinline__ void async_fence() {
#if HAVE_ASYNC_LDS
#if __has_builtin(__builtin_amdgcn_s_wait_asynccnt)
    __builtin_amdgcn_s_wait_asynccnt(0);
#else
    asm volatile("s_wait_asynccnt 0" ::: "memory");
#endif
#endif
}

// ---------------- K1: h0 = x @ w0  [4096,128]@[128,16] -> [4096,16] ----------
__global__ void k_h0(const float* __restrict__ x, const float* __restrict__ w0,
                     float* __restrict__ h0) {
    int t = blockIdx.x * blockDim.x + threadIdx.x;   // 65536 threads
    int row = t >> 4, d = t & 15;
    const float* xr = x + (size_t)row * F_IN;
    float acc = 0.f;
#pragma unroll 8
    for (int k = 0; k < F_IN; ++k) acc = fmaf(xr[k], w0[k * U0 + d], acc);
    h0[row * U0 + d] = acc;
}

// ---------------- K1b: a1_0 = h0@aw1_0, a2_0 = h0@aw2_0  -> [4096,4] ---------
__global__ void k_att0(const float* __restrict__ h0,
                       const float* __restrict__ aw1, const float* __restrict__ aw2,
                       float* __restrict__ a1, float* __restrict__ a2) {
    int t = blockIdx.x * blockDim.x + threadIdx.x;   // 16384 threads
    int row = t >> 2, hh = t & 3;
    float s1 = 0.f, s2 = 0.f;
#pragma unroll
    for (int d = 0; d < U0; ++d) {
        float hv = h0[row * U0 + d];
        s1 = fmaf(hv, aw1[d * H0 + hh], s1);
        s2 = fmaf(hv, aw2[d * H0 + hh], s2);
    }
    a1[row * H0 + hh] = s1;
    a2[row * H0 + hh] = s2;
}

// ---------------- K2: layer-0 attention aggregation (WMMA) -------------------
// block = 128 threads = 4 waves, one wave per head; block owns a 16-row tile.
__global__ void __launch_bounds__(128) k_gat0(const float* __restrict__ adj,
                                              const float* __restrict__ h0,
                                              const float* __restrict__ a1p,
                                              const float* __restrict__ a2p,
                                              float* __restrict__ out0) {
    __shared__ __align__(16) float    s_adj[16 * 36];   // 16x32 tile, pad 36
    __shared__ __align__(16) unsigned s_h[32 * 10];     // 32x16 bf16, stride 10 dw
    __shared__ __align__(16) float    s_a2[32 * 4];     // a2 tile [32 j][4 heads]

    const int tid  = threadIdx.x;
    const int hh   = tid >> 5;            // wave index == head
    const int lane = tid & 31;
    const int hi   = lane >> 4;
    const int M    = lane & 15;
    const int i0   = blockIdx.x * 16;

    const float a1v = a1p[(i0 + M) * H0 + hh];

    v8f   acc  = {};
    float wsum = 0.f;

    const int arow = tid >> 3, aquad = tid & 7;  // adj loader: 16 rows x 8 quads
    const int jr   = tid >> 2, part  = tid & 3;  // h0 loader: 32 rows x 4 quads

    for (int jt = 0; jt < N_NODES / 32; ++jt) {
        const int j0 = jt * 32;
        __syncthreads();
        // stage adj 16x32 f32 tile (async direct-to-LDS, fully coalesced b128)
        stage16(adj + (size_t)(i0 + arow) * N_NODES + j0 + aquad * 4,
                s_adj + arow * 36 + aquad * 4);
        // stage h0 rows j0..j0+31 as bf16 [32][16]
        float4 hv = *(const float4*)(h0 + (size_t)(j0 + jr) * U0 + part * 4);
        s_h[jr * 10 + part * 2]     = pack2bf(hv.x, hv.y);
        s_h[jr * 10 + part * 2 + 1] = pack2bf(hv.z, hv.w);
        // stage a2 tile
        if (tid < 32)
            *(float4*)(s_a2 + tid * 4) = *(const float4*)(a2p + (size_t)(j0 + tid) * H0);
        // prefetch next adj tile (clamped; no divergence)
        int jn = (jt + 1 < N_NODES / 32) ? (j0 + 32) : j0;
        __builtin_prefetch(adj + (size_t)(i0 + arow) * N_NODES + jn + aquad * 4, 0, 1);
        async_fence();
        __syncthreads();

        // A fragment: w[i,j] = exp(adj * relu(a1_i + a2_j)) as bf16,
        // built directly in the CDNA5 16x32 bf16 A layout.
        FragBF A;
#pragma unroll
        for (int d = 0; d < 8; ++d) {
            int k0 = ((d & 3) * 2) + ((d & 4) ? 16 : 0) + hi * 8;
            float c0 = fmaxf(a1v + s_a2[(k0)     * 4 + hh], 0.f);
            float c1 = fmaxf(a1v + s_a2[(k0 + 1) * 4 + hh], 0.f);
            float w0v = __expf(s_adj[M * 36 + k0]     * c0);
            float w1v = __expf(s_adj[M * 36 + k0 + 1] * c1);
            wsum += w0v + w1v;
            A.u[d] = pack2bf(w0v, w1v);
        }
        // B fragment: h0 tile column N = lane & 15, K in table order
        FragBF B;
        const unsigned short* sh = (const unsigned short*)s_h;
#pragma unroll
        for (int d = 0; d < 8; ++d) {
            int k0 = ((d & 3) * 2) + ((d & 4) ? 16 : 0) + hi * 8;
            B.u[d] = (unsigned)sh[k0 * 20 + M] | ((unsigned)sh[(k0 + 1) * 20 + M] << 16);
        }

        acc = __builtin_amdgcn_wmma_f32_16x16x32_bf16(false, A.v, false, B.v,
                                                      (short)0, acc, false, false);
    }

    // softmax denominator per row: lane L and L+16 hold disjoint K halves
    float den = wsum + __shfl_xor(wsum, 16);
#pragma unroll
    for (int r = 0; r < 8; ++r) {
        int Mc = r + hi * 8;                      // C row for this vgpr
        float dr = __shfl(den, Mc);               // lane Mc holds row Mc's sum
        float v  = fmaxf(acc[r] / dr, 0.f);       // relu activation
        out0[(size_t)(i0 + Mc) * 64 + hh * 16 + (lane & 15)] = v;
    }
}

// ---------------- K3: h1 = out0 @ w1, a1_1, a2_1 -----------------------------
__global__ void k_h1(const float* __restrict__ out0, const float* __restrict__ w1,
                     const float* __restrict__ aw1, const float* __restrict__ aw2,
                     float* __restrict__ h1, float* __restrict__ a1,
                     float* __restrict__ a2) {
    int row = blockIdx.x * blockDim.x + threadIdx.x;  // 4096 threads
    float acc = 0.f;
#pragma unroll 8
    for (int c = 0; c < 64; ++c) acc = fmaf(out0[(size_t)row * 64 + c], w1[c], acc);
    h1[row] = acc;
    a1[row] = acc * aw1[0];
    a2[row] = acc * aw2[0];
}

// ---------------- K4: layer-1 attention + sigmoid (WMMA) ---------------------
// B packs [h1 | ones | 0...] so one WMMA yields numerator (col0) and
// softmax denominator (col1) together.
__global__ void __launch_bounds__(32) k_gat1(const float* __restrict__ adj,
                                             const float* __restrict__ h1,
                                             const float* __restrict__ a1p,
                                             const float* __restrict__ a2p,
                                             float* __restrict__ out) {
    __shared__ __align__(16) float s_adj[16 * 36];
    __shared__ float s_a2[32];

    const int lane = threadIdx.x;
    const int hi   = lane >> 4;
    const int M    = lane & 15;
    const int i0   = blockIdx.x * 16;

    const float a1v = a1p[i0 + M];
    v8f acc = {};

    FragBF B;
#pragma unroll
    for (int d = 1; d < 8; ++d) B.u[d] = 0u;

    for (int jt = 0; jt < N_NODES / 32; ++jt) {
        const int j0 = jt * 32;
        __syncthreads();
#pragma unroll
        for (int q = 0; q < 4; ++q) {
            int idx = lane + 32 * q;
            int arow = idx >> 3, aquad = idx & 7;
            stage16(adj + (size_t)(i0 + arow) * N_NODES + j0 + aquad * 4,
                    s_adj + arow * 36 + aquad * 4);
        }
        s_a2[lane] = a2p[j0 + lane];
        float hv = h1[j0 + lane];
        async_fence();
        __syncthreads();

        FragBF A;
#pragma unroll
        for (int d = 0; d < 8; ++d) {
            int k0 = ((d & 3) * 2) + ((d & 4) ? 16 : 0) + hi * 8;
            float c0 = fmaxf(a1v + s_a2[k0],     0.f);
            float c1 = fmaxf(a1v + s_a2[k0 + 1], 0.f);
            A.u[d] = pack2bf(__expf(s_adj[M * 36 + k0]     * c0),
                             __expf(s_adj[M * 36 + k0 + 1] * c1));
        }
#pragma unroll
        for (int d = 0; d < 8; ++d) {             // col0 = h1[j], col1 = 1
            int k0 = ((d & 3) * 2) + ((d & 4) ? 16 : 0) + hi * 8;
            float p = __shfl(hv, k0), q = __shfl(hv, k0 + 1);
            B.u[d] = (M == 0) ? pack2bf(p, q) : ((M == 1) ? pack2bf(1.0f, 1.0f) : 0u);
        }

        acc = __builtin_amdgcn_wmma_f32_16x16x32_bf16(false, A.v, false, B.v,
                                                      (short)0, acc, false, false);
    }

    // C: vgpr r -> row r + 8*hi ; N=0 lanes {0,16} = num, N=1 lanes {1,17} = den
    float num = 0.f, den = 1.f;
#pragma unroll
    for (int r = 0; r < 8; ++r) {
        float nA = __shfl(acc[r], 0);
        float dA = __shfl(acc[r], 1);
        float nB = __shfl(acc[r], 16);
        float dB = __shfl(acc[r], 17);
        if (lane == r)     { num = nA; den = dA; }
        if (lane == r + 8) { num = nB; den = dB; }
    }
    if (lane < 16)
        out[i0 + lane] = 1.f / (1.f + __expf(-(num / den)));
}

// ---------------- launch -----------------------------------------------------
extern "C" void kernel_launch(void* const* d_in, const int* in_sizes, int n_in,
                              void* d_out, int out_size, void* d_ws, size_t ws_size,
                              hipStream_t stream) {
    const float* x     = (const float*)d_in[0];
    const float* adj   = (const float*)d_in[1];
    const float* w0    = (const float*)d_in[2];
    const float* aw1_0 = (const float*)d_in[3];
    const float* aw2_0 = (const float*)d_in[4];
    const float* w1    = (const float*)d_in[5];
    const float* aw1_1 = (const float*)d_in[6];
    const float* aw2_1 = (const float*)d_in[7];

    float* ws   = (float*)d_ws;
    float* h0   = ws;                 // 4096*16
    float* a1_0 = h0   + 65536;       // 4096*4
    float* a2_0 = a1_0 + 16384;       // 4096*4
    float* out0 = a2_0 + 16384;       // 4096*64
    float* h1   = out0 + 262144;      // 4096
    float* a1_1 = h1   + 4096;        // 4096
    float* a2_1 = a1_1 + 4096;        // 4096  (total ~1.43 MB)

    float* out = (float*)d_out;

    k_h0  <<<256, 256, 0, stream>>>(x, w0, h0);
    k_att0<<< 64, 256, 0, stream>>>(h0, aw1_0, aw2_0, a1_0, a2_0);
    k_gat0<<<256, 128, 0, stream>>>(adj, h0, a1_0, a2_0, out0);
    k_h1  <<< 16, 256, 0, stream>>>(out0, w1, aw1_1, aw2_1, h1, a1_1, a2_1);
    k_gat1<<<256,  32, 0, stream>>>(adj, h1, a1_1, a2_1, out);
}
